// MSDeformAttn_7868380086411
// MI455X (gfx1250) — matrix operand order repaired and checked by reading, not compiled, not run on hardware
//
#include <hip/hip_runtime.h>
#include <hip/hip_bf16.h>

typedef __attribute__((ext_vector_type(16))) _Float16 v16h;
typedef __attribute__((ext_vector_type(8)))  float    v8f;

constexpr int NB  = 2;        // batch
constexpr int LQ  = 16320;    // query length
constexpr int LEN = 16320;    // input length
constexpr int DM  = 256;      // d_model
constexpr int NH  = 8;        // heads
constexpr int DH  = 32;       // head dim
constexpr int NL  = 4;        // levels
constexpr int NP  = 4;        // points
constexpr int NF  = 3;        // frames

// ---------------------------------------------------------------------------
// Pack a row-major fp32 weight matrix W[K][Nout] into the CDNA5 WMMA B-fragment
// layout (f16), so the GEMM loads each per-lane fragment as one contiguous 32B.
// k order used here (same as A) per 32x16 (KxN) tile: lane l -> col n = l&15; lanes<16 hold K 0..15,
// lanes>=16 hold K 16..31; vgpr i holds halves {K=2i, K=2i+1}.
// Packed order: tile = kt*(Nout/16)+nt; addr = (tile*32 + lane)*16 halves.
// ---------------------------------------------------------------------------
__global__ void pack_w_f16(const float* __restrict__ W, _Float16* __restrict__ out,
                           int K, int Nout) {
  int tid  = blockIdx.x * blockDim.x + threadIdx.x;
  int lane = tid & 31;
  int tile = tid >> 5;
  int ntn  = Nout >> 4;
  int ntk  = K >> 5;
  if (tile >= ntn * ntk) return;
  int nt = tile % ntn;
  int kt = tile / ntn;
  int nn = nt * 16 + (lane & 15);
  int kb = kt * 32 + (lane >> 4) * 16;
  _Float16* o = out + ((size_t)tile * 32 + lane) * 16;
#pragma unroll
  for (int i = 0; i < 8; ++i) {
    o[2*i]   = (_Float16)W[(size_t)(kb + 2*i    ) * Nout + nn];
    o[2*i+1] = (_Float16)W[(size_t)(kb + 2*i + 1) * Nout + nn];
  }
}

// ---------------------------------------------------------------------------
// C = A(f32->f16) @ Bpacked + bias.  One wave per 16(M) x 64(N) strip:
// 4 f32 accumulators, A fragment reused 4x per K-step of 32.
// A frag layout (16x32 f16): lane: m = lane&15, hi = lane>>4;
//   vgpr i holds halves {k, k+1} with k = hi*16 + 2*i.
// ---------------------------------------------------------------------------
__global__ void wmma_gemm_bias(const float* __restrict__ A,
                               const _Float16* __restrict__ Bp,
                               const float* __restrict__ bias,
                               float* __restrict__ C,
                               int M, int K, int Nout) {
  int lane = threadIdx.x & 31;
  int w    = (blockIdx.x * blockDim.x + threadIdx.x) >> 5;
  int ngroups = Nout >> 6;            // 64-wide N strips
  int mt = w / ngroups;
  int ng = w % ngroups;
  if (mt * 16 >= M) return;
  int ntn = Nout >> 4;
  int hi  = lane >> 4;
  const float* arow = A + (size_t)(mt * 16 + (lane & 15)) * K;
  const v16h*  bb   = (const v16h*)Bp;   // one v16h = one lane fragment (32B)

  v8f c0 = {}, c1 = {}, c2 = {}, c3 = {};
  for (int kt = 0; kt < (K >> 5); ++kt) {
    v16h a;
#pragma unroll
    for (int i = 0; i < 8; ++i) {
      int kk = kt * 32 + hi * 16 + 2 * i;
      a[2*i]   = (_Float16)arow[kk];
      a[2*i+1] = (_Float16)arow[kk + 1];
    }
    size_t tb = (size_t)kt * ntn + (size_t)ng * 4;
    v16h b0 = bb[(tb + 0) * 32 + lane];
    v16h b1 = bb[(tb + 1) * 32 + lane];
    v16h b2 = bb[(tb + 2) * 32 + lane];
    v16h b3 = bb[(tb + 3) * 32 + lane];
    c0 = __builtin_amdgcn_wmma_f32_16x16x32_f16(false, a, false, b0, (short)0, c0, false, false);
    c1 = __builtin_amdgcn_wmma_f32_16x16x32_f16(false, a, false, b1, (short)0, c1, false, false);
    c2 = __builtin_amdgcn_wmma_f32_16x16x32_f16(false, a, false, b2, (short)0, c2, false, false);
    c3 = __builtin_amdgcn_wmma_f32_16x16x32_f16(false, a, false, b3, (short)0, c3, false, false);
  }

  // C/D layout: lane col = lane&15; vgpr i -> row = i + 8*hi
  int mrow0 = mt * 16 + 8 * hi;
  v8f cc[4] = {c0, c1, c2, c3};
#pragma unroll
  for (int j = 0; j < 4; ++j) {
    int n   = ng * 64 + j * 16 + (lane & 15);
    float bv = bias[n];
    v8f c = cc[j];
#pragma unroll
    for (int i = 0; i < 8; ++i)
      C[(size_t)(mrow0 + i) * Nout + n] = c[i] + bv;
  }
}

// ---------------------------------------------------------------------------
// Deformable sampling: one wave per (n, q, h); 32 lanes = the 32 head channels,
// so every bilinear tap is a coalesced 128B read of value[..., h, 0:32].
// Softmax over the 16 (L*P) logits of `to` (temporal offset) and `aw`
// (attention weight) computed per-lane from uniform-address broadcast loads.
// ---------------------------------------------------------------------------
__global__ void ms_deform_sample(const float* __restrict__ value,   // (NB,LEN,NH,DH)
                                 const float* __restrict__ so_buf,  // (NB,LQ,NH,L,P,2)
                                 const float* __restrict__ to_buf,  // (NB,LQ,NH,16) logits
                                 const float* __restrict__ aw_buf,  // (NB,LQ,NH,16) logits
                                 const float* __restrict__ ref,     // (NB,LQ,NL,3)
                                 const int*   __restrict__ ss,      // (NL,2)
                                 const int*   __restrict__ ls,      // (NL)
                                 float* __restrict__ out) {         // (NB,LQ,DM)
  int lane = threadIdx.x & 31;
  int w    = (blockIdx.x * blockDim.x + threadIdx.x) >> 5;
  if (w >= NB * LQ * NH) return;
  int h = w % NH;
  int q = (w / NH) % LQ;
  int n = w / (NH * LQ);
  size_t row = (size_t)n * LQ + q;
  size_t b16 = (row * NH + h) * 16;

  float tw[16], aw[16];
  float tmax = -1e30f, amax = -1e30f;
#pragma unroll
  for (int j = 0; j < 16; ++j) {
    tw[j] = to_buf[b16 + j];
    aw[j] = aw_buf[b16 + j];
    tmax = fmaxf(tmax, tw[j]);
    amax = fmaxf(amax, aw[j]);
  }
  float ts = 0.f, as = 0.f;
#pragma unroll
  for (int j = 0; j < 16; ++j) {
    tw[j] = __expf(tw[j] - tmax); ts += tw[j];
    aw[j] = __expf(aw[j] - amax); as += aw[j];
  }
  float tri = 1.f / ts, ari = 1.f / as;

  size_t sob = b16 * 2;
  float acc = 0.f;
  for (int l = 0; l < NL; ++l) {
    int Hh = ss[2*l], Ww = ss[2*l + 1];
    int start = ls[l];
    float rx = ref[(row * NL + l) * 3 + 0];
    float ry = ref[(row * NL + l) * 3 + 1];
    float rt = ref[(row * NL + l) * 3 + 2];
    float invW = 1.f / (float)Ww, invH = 1.f / (float)Hh;
#pragma unroll
    for (int p = 0; p < NP; ++p) {
      int j = l * NP + p;
      float lx = rx + so_buf[sob + 2*j]     * invW;
      float ly = ry + so_buf[sob + 2*j + 1] * invH;
      float lt = rt + tw[j] * tri;               // temporal offset (norm=1)
      int t = (int)rintf(lt * (float)(NF - 1));  // round-to-nearest-even
      t = t < 0 ? 0 : (t > NF - 1 ? NF - 1 : t);
      float x = lx * (float)Ww - 0.5f;
      float y = ly * (float)Hh - 0.5f;
      float x0f = floorf(x), y0f = floorf(y);
      float fx = x - x0f, fy = y - y0f;
      int x0 = (int)x0f, y0 = (int)y0f;
      float a = aw[j] * ari;
      size_t base = (size_t)n * LEN + start + (size_t)t * Hh * Ww;
#pragma unroll
      for (int dy = 0; dy < 2; ++dy) {
#pragma unroll
        for (int dx = 0; dx < 2; ++dx) {
          int xi = x0 + dx, yi = y0 + dy;
          if (xi >= 0 && xi < Ww && yi >= 0 && yi < Hh) {
            float wgt = (dx ? fx : 1.f - fx) * (dy ? fy : 1.f - fy);
            float v = value[((base + (size_t)yi * Ww + xi) * NH + h) * DH + lane];
            acc += a * wgt * v;
          }
        }
      }
    }
  }
  out[row * DM + h * DH + lane] = acc;
}

// ---------------------------------------------------------------------------
extern "C" void kernel_launch(void* const* d_in, const int* in_sizes, int n_in,
                              void* d_out, int out_size, void* d_ws, size_t ws_size,
                              hipStream_t stream) {
  const float* query = (const float*)d_in[0];
  const float* refp  = (const float*)d_in[1];
  const float* inpf  = (const float*)d_in[2];
  const int*   ss    = (const int*)d_in[3];
  const int*   ls    = (const int*)d_in[4];
  const float* W_so  = (const float*)d_in[5];
  const float* b_so  = (const float*)d_in[6];
  const float* W_to  = (const float*)d_in[7];
  const float* b_to  = (const float*)d_in[8];
  const float* W_a   = (const float*)d_in[9];
  const float* b_a   = (const float*)d_in[10];
  const float* W_v   = (const float*)d_in[11];
  const float* b_v   = (const float*)d_in[12];
  const float* W_o   = (const float*)d_in[13];
  const float* b_o   = (const float*)d_in[14];
  float* out = (float*)d_out;

  // workspace carve-out (~134 MB)
  char* ws = (char*)d_ws;
  size_t off = 0;
  auto carve = [&](size_t bytes) -> void* {
    void* p = ws + off;
    off = (off + bytes + 255) & ~(size_t)255;
    return p;
  };
  float* value_buf = (float*)carve((size_t)NB * LEN * DM  * 4);
  float* so_buf    = (float*)carve((size_t)NB * LQ  * DM  * 4);
  float* to_buf    = (float*)carve((size_t)NB * LQ  * 128 * 4);
  float* aw_buf    = (float*)carve((size_t)NB * LQ  * 128 * 4);
  float* tmp_buf   = (float*)carve((size_t)NB * LQ  * DM  * 4);
  _Float16* pWv  = (_Float16*)carve((size_t)DM * DM  * 2);
  _Float16* pWso = (_Float16*)carve((size_t)DM * DM  * 2);
  _Float16* pWto = (_Float16*)carve((size_t)DM * 128 * 2);
  _Float16* pWa  = (_Float16*)carve((size_t)DM * 128 * 2);
  _Float16* pWo  = (_Float16*)carve((size_t)DM * DM  * 2);

  // 1) pack weights into WMMA B-fragment layout (f16)
  auto pack = [&](const float* W, _Float16* P, int K, int Nout) {
    int threads = (K >> 5) * (Nout >> 4) * 32;
    pack_w_f16<<<(threads + 255) / 256, 256, 0, stream>>>(W, P, K, Nout);
  };
  pack(W_v,  pWv,  DM, DM);
  pack(W_so, pWso, DM, DM);
  pack(W_to, pWto, DM, 128);
  pack(W_a,  pWa,  DM, 128);
  pack(W_o,  pWo,  DM, DM);

  // 2) projection GEMMs (WMMA)
  auto gemm = [&](const float* A, const _Float16* B, const float* bias,
                  float* C, int M, int K, int Nout) {
    int waves  = (M / 16) * (Nout / 64);
    int blocks = (waves * 32 + 255) / 256;
    wmma_gemm_bias<<<blocks, 256, 0, stream>>>(A, B, bias, C, M, K, Nout);
  };
  gemm(inpf,  pWv,  b_v,  value_buf, NB * LEN, DM, DM);   // value projection
  gemm(query, pWso, b_so, so_buf,    NB * LQ,  DM, DM);   // sampling offsets
  gemm(query, pWto, b_to, to_buf,    NB * LQ,  DM, 128);  // temporal logits
  gemm(query, pWa,  b_a,  aw_buf,    NB * LQ,  DM, 128);  // attention logits

  // 3) deformable bilinear sampling (gather-bound, coalesced per head)
  {
    int waves  = NB * LQ * NH;
    int blocks = (waves * 32 + 255) / 256;
    ms_deform_sample<<<blocks, 256, 0, stream>>>(value_buf, so_buf, to_buf,
                                                 aw_buf, refp, ss, ls, tmp_buf);
  }

  // 4) output projection (WMMA)
  gemm(tmp_buf, pWo, b_o, out, NB * LQ, DM, DM);
}
